// FeatureSpaceSCTABaseline_34342558499430
// MI455X (gfx1250) — compile-verified, matrix operand order checked by reading, not run on hardware
//
#include <hip/hip_runtime.h>
#include <hip/hip_bf16.h>

// ---------------------------------------------------------------------------
// MI455X (gfx1250) implementation of FeatureSpaceSCTA forward.
//
// Shapes (all hardcoded): B=4, T=8, H=W=256, P=4 -> h=w=64, FD=HID=128,
// L=64, SD=64.  Dominant cost: 32 ConvGRU 3x3 convs with C_in=256,
// C_out in {256,128} on 4x(64x64) images (~460 GFLOP) -> implicit-GEMM
// with v_wmma_f32_16x16x32_f16 (f16 operands, f32 accumulate).
// Activation staging uses GLOBAL_LOAD_ASYNC_TO_LDS_B128 (ASYNCcnt path)
// so the copy overlaps with WMMA + weight-fragment traffic.
// ---------------------------------------------------------------------------

typedef _Float16 v16h __attribute__((ext_vector_type(16)));
typedef _Float16 v8h  __attribute__((ext_vector_type(8)));
typedef float    v8f  __attribute__((ext_vector_type(8)));

#define SHUF16(lo, hi) __builtin_shufflevector((lo), (hi), 0,1,2,3,4,5,6,7,8,9,10,11,12,13,14,15)

// Per-lane K index inside a 32-wide K block for the 16-bit WMMA A/B layout:
// lanes 0-15: K = 0..7 (VGPR0-3) and 16..23 (VGPR4-7); lanes 16-31: +8.
__device__ __forceinline__ int kmap(int j, int half) {
  return (j < 8) ? (half * 8 + j) : (16 + half * 8 + (j - 8));
}

__device__ __forceinline__ int lattice_idx(float cx, float cy) {
  int ix = (int)(cx * 64.0f); ix = ix < 0 ? 0 : (ix > 63 ? 63 : ix);
  int iy = (int)(cy * 64.0f); iy = iy < 0 ? 0 : (iy > 63 ? 63 : iy);
  return iy * 64 + ix;
}

__device__ __forceinline__ float geluf(float x) {
  float x3 = x * x * x;
  return 0.5f * x * (1.0f + tanhf(0.7978845608f * (x + 0.044715f * x3)));
}

__device__ __forceinline__ float sigf(float x) { return 1.0f / (1.0f + expf(-x)); }

// Async memory->LDS 16B copy (gfx1250).  ldsoff = raw LDS byte offset
// (low 32 bits of the generic shared-memory address, per the LDS aperture
// rule ADDR[31:0] -> LDS offset).  Tracked on ASYNCcnt.
__device__ __forceinline__ void async_ld_b128(void* lds_dst, const void* gsrc) {
  unsigned ldsoff = (unsigned)(size_t)lds_dst;
  unsigned long long ga = (unsigned long long)(size_t)gsrc;
  asm volatile("global_load_async_to_lds_b128 %0, %1, off"
               :: "v"(ldsoff), "v"(ga) : "memory");
}
__device__ __forceinline__ void wait_async0() {
  asm volatile("s_wait_asynccnt 0x0" ::: "memory");
}

// ---------------------------------------------------------------------------
// Zero helper (d_out/d_ws are poisoned; re-init every call, deterministic).
// ---------------------------------------------------------------------------
__global__ void k_zero(float* __restrict__ p, size_t n) {
  size_t i = (size_t)blockIdx.x * blockDim.x + threadIdx.x;
  size_t stride = (size_t)gridDim.x * blockDim.x;
  for (; i < n; i += stride) p[i] = 0.0f;
}

// ---------------------------------------------------------------------------
// Weight swizzle: OIHW f32 (Cout, 256, 3, 3) -> per-lane B-fragment f16:
// frag[((nt*9 + tap)*8 + kc)*512 + lane*16 + j], n = lane&15, co = nt*16+n,
// ci = kc*32 + kmap(j, lane>>4).
// ---------------------------------------------------------------------------
__global__ void k_prep_w3(const float* __restrict__ w, _Float16* __restrict__ frag, int NT) {
  int f = blockIdx.x * 256 + threadIdx.x;
  int total = NT * 9 * 8 * 512;
  if (f >= total) return;
  int j = f & 15, lane = (f >> 4) & 31, kc = (f >> 9) & 7;
  int rest = f >> 12, tap = rest % 9, nt = rest / 9;
  int n0 = lane & 15, half = lane >> 4;
  int ci = kc * 32 + kmap(j, half);
  int co = nt * 16 + n0;
  frag[f] = (_Float16)w[((size_t)(co * 256 + ci)) * 9 + tap];
}

// Patch weights (128,3,4,4): K = c*16 + py*4 + px (48 real, padded to 64).
__global__ void k_prep_wp(const float* __restrict__ pw, _Float16* __restrict__ frag) {
  int f = blockIdx.x * 256 + threadIdx.x;  // 8*2*512 = 8192
  if (f >= 8192) return;
  int j = f & 15, lane = (f >> 4) & 31, kc = (f >> 9) & 1, nt = f >> 10;
  int n0 = lane & 15, half = lane >> 4;
  int k = kc * 32 + kmap(j, half);
  int co = nt * 16 + n0;
  float v = 0.0f;
  if (k < 48) {
    int c = k >> 4, pos = k & 15;
    v = pw[(co * 3 + c) * 16 + pos];
  }
  frag[f] = (_Float16)v;
}

// ---------------------------------------------------------------------------
// Patch embed (stride-4 4x4 conv, K=48->64) + GELU, fused with the
// scatter-mean splat: atomicAdd into world sums, one block per (bt, out-row).
// 8 waves: wave&3 = M subtile (16 patches), 4 N-tiles per wave, 2 WMMAs each.
// (Staging here converts f32->f16, so it stays on the VALU path.)
// ---------------------------------------------------------------------------
__global__ __launch_bounds__(256)
void k_patch(const float* __restrict__ obs, const float* __restrict__ scoord,
             const _Float16* __restrict__ wp, const float* __restrict__ pb,
             float* __restrict__ world) {
  int blk = blockIdx.x;            // bt*64 + y
  int bt = blk >> 6, y = blk & 63;
  __shared__ __align__(16) _Float16 sA[64 * 64];  // [patch][K(64)]
  __shared__ int sIdx[64];
  int tid = threadIdx.x;
  for (int i = tid; i < 4096; i += 256) {
    int patch = i >> 6, k = i & 63;
    float v = 0.0f;
    if (k < 48) {
      int c = k >> 4, pos = k & 15, py = pos >> 2, px = pos & 3;
      v = obs[(((size_t)bt * 3 + c) * 256 + (y * 4 + py)) * 256 + patch * 4 + px];
    }
    sA[i] = (_Float16)v;
  }
  if (tid < 64) {
    float cx = scoord[((size_t)bt * 2 + 0) * 4096 + y * 64 + tid];
    float cy = scoord[((size_t)bt * 2 + 1) * 4096 + y * 64 + tid];
    sIdx[tid] = lattice_idx(cx, cy);
  }
  __syncthreads();

  int wave = tid >> 5, lane = tid & 31;
  int msub = wave & 3, half = lane >> 4, n0 = lane & 15;
  int m = msub * 16 + n0;          // A row = patch index within the out row
  for (int nt = (wave >> 2); nt < 8; nt += 2) {
    v8f acc;
    float bv = pb[nt * 16 + n0];
    #pragma unroll
    for (int r = 0; r < 8; ++r) acc[r] = bv;
    #pragma unroll
    for (int kc = 0; kc < 2; ++kc) {
      const _Float16* ap = &sA[m * 64 + kc * 32 + half * 8];
      v8h a0 = *(const v8h*)ap, a1 = *(const v8h*)(ap + 16);
      v16h av = SHUF16(a0, a1);
      const _Float16* bp = wp + ((size_t)nt * 2 + kc) * 512 + lane * 16;
      v8h b0 = *(const v8h*)bp, b1 = *(const v8h*)(bp + 8);
      v16h bvv = SHUF16(b0, b1);
      acc = __builtin_amdgcn_wmma_f32_16x16x32_f16(false, av, false, bvv,
                                                   (short)0, acc, false, false);
    }
    int co = nt * 16 + n0;
    float* wdst = world + ((size_t)bt * 128 + co) * 4096;
    #pragma unroll
    for (int r = 0; r < 8; ++r) {
      int x = msub * 16 + 8 * half + r;
      atomicAdd(&wdst[sIdx[x]], geluf(acc[r]));
    }
  }
}

// One count per source pixel.
__global__ void k_occ(const float* __restrict__ scoord, float* __restrict__ occ) {
  int f = blockIdx.x * 256 + threadIdx.x;  // 32*4096
  int bt = f >> 12, pix = f & 4095;
  float cx = scoord[((size_t)bt * 2 + 0) * 4096 + pix];
  float cy = scoord[((size_t)bt * 2 + 1) * 4096 + pix];
  atomicAdd(&occ[(size_t)bt * 4096 + lattice_idx(cx, cy)], 1.0f);
}

__global__ void k_norm_world(float* __restrict__ world, const float* __restrict__ occ) {
  size_t f = (size_t)blockIdx.x * 256 + threadIdx.x;  // 32*128*4096
  int pix = (int)(f & 4095);
  int bt = (int)(f >> 19);
  world[f] /= fmaxf(occ[(size_t)bt * 4096 + pix], 1.0f);
}

// ---------------------------------------------------------------------------
// 3x3 SAME conv, C_in=256 -> C_out=NT*16, implicit GEMM on WMMA.
// One block per (b, output row y).  512 threads = 16 waves.
// Input xin: f16 pixel-major (b, 4096 pix, 256 ch).  LDS stages rows
// y-1..y+1 in two 128-channel chunks (48KB each) via async LDS loads;
// accumulators persist across chunks.  Weights pre-swizzled by k_prep_w3.
// ---------------------------------------------------------------------------
template <int NT>
__global__ __launch_bounds__(512)
void k_conv3x3(const _Float16* __restrict__ xin, const _Float16* __restrict__ wf,
               const float* __restrict__ bias, float* __restrict__ out) {
  constexpr int NITER = NT / 4;
  int blk = blockIdx.x;              // b*64 + y
  int b = blk >> 6, y = blk & 63;
  __shared__ __align__(16) _Float16 sIn[3 * 64 * 128];  // 48 KB
  int tid = threadIdx.x, wave = tid >> 5, lane = tid & 31;
  int msub = wave & 3, half = lane >> 4, n0 = lane & 15;
  int ntbase = wave >> 2;

  v8f acc[NITER];
  #pragma unroll
  for (int it = 0; it < NITER; ++it) {
    int nt = ntbase + it * 4;
    float bv = bias[nt * 16 + n0];
    #pragma unroll
    for (int r = 0; r < 8; ++r) acc[it][r] = bv;
  }
  int xo = msub * 16 + n0;           // A row = output x within row y

  for (int cblk = 0; cblk < 2; ++cblk) {
    __syncthreads();
    // Stage 3 rows x 64 px x 128 ch (f16): async global->LDS for in-range
    // rows (ASYNCcnt), plain LDS zero-stores for the padded halo rows.
    for (int i = tid; i < 3 * 64 * 128 / 8; i += 512) {
      int row = i / (64 * 128 / 8);
      int off = (i % (64 * 128 / 8)) * 8;   // halfs within the row slab
      int ry = y - 1 + row;
      _Float16* dst = &sIn[row * 64 * 128 + off];
      if ((unsigned)ry < 64u) {
        int px = off >> 7, cc = off & 127;
        const _Float16* src =
            &xin[(((size_t)b * 4096) + ry * 64 + px) * 256 + cblk * 128 + cc];
        async_ld_b128(dst, src);
      } else {
        *(v8h*)dst = (v8h){};
      }
    }
    wait_async0();        // per-wave: async copies landed in LDS
    __syncthreads();      // all waves' stores visible

    #pragma unroll
    for (int it = 0; it < NITER; ++it) {
      int nt = ntbase + it * 4;
      const _Float16* wb = wf + (size_t)nt * 9 * 8 * 512;
      // Pull next N-tile's fragments toward the WGP while this tile computes.
      if (it + 1 < NITER)
        __builtin_prefetch(wf + (size_t)(ntbase + (it + 1) * 4) * 9 * 8 * 512, 0, 0);
      #pragma unroll
      for (int tap = 0; tap < 9; ++tap) {
        int ky = tap / 3, dx = tap % 3 - 1;
        int xi = xo + dx;
        bool valid = (unsigned)xi < 64u;
        const _Float16* ab = &sIn[(ky * 64 + xi) * 128 + half * 8];
        #pragma unroll
        for (int kcl = 0; kcl < 4; ++kcl) {
          v8h a0 = {}, a1 = {};
          if (valid) {
            a0 = *(const v8h*)(ab + kcl * 32);
            a1 = *(const v8h*)(ab + kcl * 32 + 16);
          }
          v16h av = SHUF16(a0, a1);
          const _Float16* bp = wb + ((size_t)tap * 8 + cblk * 4 + kcl) * 512 + lane * 16;
          v8h b0 = *(const v8h*)bp, b1 = *(const v8h*)(bp + 8);
          v16h bvv = SHUF16(b0, b1);
          acc[it] = __builtin_amdgcn_wmma_f32_16x16x32_f16(false, av, false, bvv,
                                                           (short)0, acc[it], false, false);
        }
      }
    }
  }

  #pragma unroll
  for (int it = 0; it < NITER; ++it) {
    int nt = ntbase + it * 4;
    int co = nt * 16 + n0;
    float* op = out + (((size_t)b * NT * 16) + co) * 4096 + y * 64 + msub * 16 + 8 * half;
    #pragma unroll
    for (int r = 0; r < 8; ++r) op[r] = acc[it][r];
  }
}

// ---------------------------------------------------------------------------
// GRU glue kernels (memory-bound, plain VALU).
// ---------------------------------------------------------------------------
// xh[b][pix][ci]: ci<128 from x (f32, (B,T,128,4096) at step t), else h.
__global__ void k_pack_f32(const float* __restrict__ x, const float* __restrict__ h,
                           _Float16* __restrict__ xh, int t) {
  size_t f = (size_t)blockIdx.x * 256 + threadIdx.x;   // 4*256*4096
  int pix = (int)(f & 4095), ci = (int)((f >> 12) & 255), b = (int)(f >> 20);
  float v = (ci < 128)
      ? x[(((size_t)b * 8 + t) * 128 + ci) * 4096 + pix]
      : h[((size_t)b * 128 + (ci - 128)) * 4096 + pix];
  xh[((size_t)b * 4096 + pix) * 256 + ci] = (_Float16)v;
}

__global__ void k_pack_f16(const _Float16* __restrict__ x, const float* __restrict__ h,
                           _Float16* __restrict__ xh, int t) {
  size_t f = (size_t)blockIdx.x * 256 + threadIdx.x;
  int pix = (int)(f & 4095), ci = (int)((f >> 12) & 255), b = (int)(f >> 20);
  float v = (ci < 128)
      ? (float)x[(((size_t)b * 8 + t) * 128 + ci) * 4096 + pix]
      : h[((size_t)b * 128 + (ci - 128)) * 4096 + pix];
  xh[((size_t)b * 4096 + pix) * 256 + ci] = (_Float16)v;
}

// z = sigmoid(zr[:128]); r = sigmoid(zr[128:]); xh[...,128+c] = f16(r*h)
__global__ void k_zr_act(const float* __restrict__ zr, const float* __restrict__ h,
                         float* __restrict__ z, _Float16* __restrict__ xh) {
  size_t f = (size_t)blockIdx.x * 256 + threadIdx.x;   // 4*128*4096
  int pix = (int)(f & 4095), c = (int)((f >> 12) & 127), b = (int)(f >> 19);
  float zv = sigf(zr[((size_t)b * 256 + c) * 4096 + pix]);
  float rv = sigf(zr[((size_t)b * 256 + 128 + c) * 4096 + pix]);
  z[f] = zv;
  xh[((size_t)b * 4096 + pix) * 256 + 128 + c] = (_Float16)(rv * h[f]);
}

// h = (1-z)*h + z*tanh(htil); hs[b,t,c,pix] = f16(h)
__global__ void k_update(const float* __restrict__ z, const float* __restrict__ htil,
                         float* __restrict__ h, _Float16* __restrict__ hs, int t) {
  size_t f = (size_t)blockIdx.x * 256 + threadIdx.x;   // 4*128*4096
  int pix = (int)(f & 4095), c = (int)((f >> 12) & 127), b = (int)(f >> 19);
  float hn = (1.0f - z[f]) * h[f] + z[f] * tanhf(htil[f]);
  h[f] = hn;
  hs[(((size_t)b * 8 + t) * 128 + c) * 4096 + pix] = (_Float16)hn;
}

// ---------------------------------------------------------------------------
// Sensor embedding: stats = [mean coord x, mean coord y, scale] -> 3->64
// GELU -> 64->64.  One block per batch; writes cond[b, colOff..colOff+64).
// ---------------------------------------------------------------------------
__global__ void k_sensor(const float* __restrict__ coord, const float* __restrict__ scale,
                         const float* __restrict__ w1, const float* __restrict__ b1,
                         const float* __restrict__ w2, const float* __restrict__ b2,
                         float* __restrict__ cond, int colOff) {
  int b = blockIdx.x, tid = threadIdx.x;
  __shared__ float red[256];
  __shared__ float stats[3];
  __shared__ float hdn[64];
  for (int c = 0; c < 2; ++c) {
    float s = 0.0f;
    for (int i = tid; i < 8 * 4096; i += 256) {
      int t = i >> 12, pix = i & 4095;
      s += coord[(((size_t)b * 8 + t) * 2 + c) * 4096 + pix];
    }
    red[tid] = s; __syncthreads();
    for (int o = 128; o; o >>= 1) { if (tid < o) red[tid] += red[tid + o]; __syncthreads(); }
    if (tid == 0) stats[c] = red[0] / 32768.0f;
    __syncthreads();
  }
  if (tid == 0) stats[2] = scale[b];
  __syncthreads();
  if (tid < 64) {
    float a = b1[tid];
    for (int i = 0; i < 3; ++i) a += stats[i] * w1[i * 64 + tid];
    hdn[tid] = geluf(a);
  }
  __syncthreads();
  if (tid < 64) {
    float a = b2[tid];
    for (int k = 0; k < 64; ++k) a += hdn[k] * w2[k * 64 + tid];
    cond[b * 128 + colOff + tid] = a;
  }
}

// mod MLP: cond(4,128) -> relu(128) -> 256; gb[b][0:128]=gamma, [128:256]=beta.
__global__ void k_mod(const float* __restrict__ cond,
                      const float* __restrict__ w1, const float* __restrict__ b1,
                      const float* __restrict__ w2, const float* __restrict__ b2,
                      float* __restrict__ gb) {
  int b = blockIdx.x, tid = threadIdx.x;   // 256 threads
  __shared__ float hid[128];
  if (tid < 128) {
    float a = b1[tid];
    for (int k = 0; k < 128; ++k) a += cond[b * 128 + k] * w1[k * 128 + tid];
    hid[tid] = fmaxf(a, 0.0f);
  }
  __syncthreads();
  float a = b2[tid];
  for (int j = 0; j < 128; ++j) a += hid[j] * w2[j * 256 + tid];
  gb[b * 256 + tid] = a;
}

// gather from lattice + FiLM -> decoded (B,T,128,64,64)
__global__ void k_gather(const _Float16* __restrict__ hs1, const float* __restrict__ tcoord,
                         const float* __restrict__ gb, float* __restrict__ dec) {
  size_t f = (size_t)blockIdx.x * 256 + threadIdx.x;   // 4*8*128*4096
  int pix = (int)(f & 4095), c = (int)((f >> 12) & 127);
  int t = (int)((f >> 19) & 7), b = (int)(f >> 22);
  float cx = tcoord[(((size_t)b * 8 + t) * 2 + 0) * 4096 + pix];
  float cy = tcoord[(((size_t)b * 8 + t) * 2 + 1) * 4096 + pix];
  int idx = lattice_idx(cx, cy);
  float g = (float)hs1[(((size_t)b * 8 + t) * 128 + c) * 4096 + idx];
  dec[f] = g * (1.0f + gb[b * 256 + c]) + gb[b * 256 + 128 + c];
}

// position head: 1-channel 3x3 conv over decoded[:, -1]
__global__ void k_logits(const float* __restrict__ dec, const float* __restrict__ w,
                         const float* __restrict__ bias, float* __restrict__ logits) {
  int f = blockIdx.x * 256 + threadIdx.x;  // 4*4096
  int b = f >> 12, pix = f & 4095, y = pix >> 6, x = pix & 63;
  float acc = bias[0];
  for (int c = 0; c < 128; ++c) {
    const float* xp = dec + (((size_t)b * 8 + 7) * 128 + c) * 4096;
    #pragma unroll
    for (int ky = 0; ky < 3; ++ky) {
      int yy = y + ky - 1;
      if ((unsigned)yy >= 64u) continue;
      #pragma unroll
      for (int kx = 0; kx < 3; ++kx) {
        int xx = x + kx - 1;
        if ((unsigned)xx >= 64u) continue;
        acc += xp[yy * 64 + xx] * w[(c * 3 + ky) * 3 + kx];
      }
    }
  }
  logits[f] = acc;
}

// softmax heatmap + soft-argmax + pooled depth MLP; writes heat and pred.
__global__ void k_head(const float* __restrict__ logits, const float* __restrict__ dec,
                       const float* __restrict__ dw1, const float* __restrict__ db1,
                       const float* __restrict__ dw2, const float* __restrict__ db2,
                       float* __restrict__ heat, float* __restrict__ pred) {
  int b = blockIdx.x, tid = threadIdx.x;  // 256 threads
  __shared__ float red[256];
  __shared__ float smax, ssum, sxn, syn;
  __shared__ float pooled[128];
  __shared__ float hid[128];
  const float* lg = logits + (size_t)b * 4096;

  float m = -1e30f;
  for (int i = tid; i < 4096; i += 256) m = fmaxf(m, lg[i]);
  red[tid] = m; __syncthreads();
  for (int o = 128; o; o >>= 1) { if (tid < o) red[tid] = fmaxf(red[tid], red[tid + o]); __syncthreads(); }
  if (!tid) smax = red[0]; __syncthreads();

  float s = 0.0f;
  for (int i = tid; i < 4096; i += 256) s += expf(lg[i] - smax);
  red[tid] = s; __syncthreads();
  for (int o = 128; o; o >>= 1) { if (tid < o) red[tid] += red[tid + o]; __syncthreads(); }
  if (!tid) ssum = red[0]; __syncthreads();

  float xn = 0.0f, yn = 0.0f;
  for (int i = tid; i < 4096; i += 256) {
    float h = expf(lg[i] - smax) / ssum;
    heat[(size_t)b * 4096 + i] = h;
    int y = i >> 6, x = i & 63;
    xn += h * (-1.0f + 2.0f * x / 63.0f);
    yn += h * (-1.0f + 2.0f * y / 63.0f);
  }
  red[tid] = xn; __syncthreads();
  for (int o = 128; o; o >>= 1) { if (tid < o) red[tid] += red[tid + o]; __syncthreads(); }
  if (!tid) sxn = red[0]; __syncthreads();
  red[tid] = yn; __syncthreads();
  for (int o = 128; o; o >>= 1) { if (tid < o) red[tid] += red[tid + o]; __syncthreads(); }
  if (!tid) syn = red[0]; __syncthreads();

  if (tid < 128) {
    const float* xp = dec + (((size_t)b * 8 + 7) * 128 + tid) * 4096;
    float s2 = 0.0f;
    for (int i = 0; i < 4096; ++i) s2 += xp[i];
    pooled[tid] = s2 / 4096.0f;
  }
  __syncthreads();
  if (tid < 128) {
    float a = db1[tid];
    for (int c = 0; c < 128; ++c) a += pooled[c] * dw1[c * 128 + tid];
    hid[tid] = fmaxf(a, 0.0f);
  }
  __syncthreads();
  red[tid] = (tid < 128) ? hid[tid] * dw2[tid] : 0.0f;
  __syncthreads();
  for (int o = 128; o; o >>= 1) { if (tid < o) red[tid] += red[tid + o]; __syncthreads(); }
  if (!tid) {
    pred[b * 3 + 0] = sxn;
    pred[b * 3 + 1] = syn;
    pred[b * 3 + 2] = red[0] + db2[0];
  }
}

// ---------------------------------------------------------------------------
// Host orchestration.
// ---------------------------------------------------------------------------
extern "C" void kernel_launch(void* const* d_in, const int* in_sizes, int n_in,
                              void* d_out, int out_size, void* d_ws, size_t ws_size,
                              hipStream_t stream) {
  (void)in_sizes; (void)n_in; (void)out_size; (void)ws_size;

  // Inputs (setup_inputs dict order; params flattened in insertion order)
  const float* obs    = (const float*)d_in[0];   // (4,8,3,256,256)
  const float* scoord = (const float*)d_in[1];   // (4,8,2,64,64)
  const float* sscale = (const float*)d_in[2];   // (4,)
  const float* tcoord = (const float*)d_in[3];   // (4,8,2,64,64)
  const float* tscale = (const float*)d_in[4];   // (4,)
  const float* patch_w = (const float*)d_in[5];
  const float* patch_b = (const float*)d_in[6];
  const float* s_w1 = (const float*)d_in[7];
  const float* s_b1 = (const float*)d_in[8];
  const float* s_w2 = (const float*)d_in[9];
  const float* s_b2 = (const float*)d_in[10];
  const float* g0_wzr = (const float*)d_in[11];
  const float* g0_bzr = (const float*)d_in[12];
  const float* g0_wh  = (const float*)d_in[13];
  const float* g0_bh  = (const float*)d_in[14];
  const float* g1_wzr = (const float*)d_in[15];
  const float* g1_bzr = (const float*)d_in[16];
  const float* g1_wh  = (const float*)d_in[17];
  const float* g1_bh  = (const float*)d_in[18];
  const float* mod_w1 = (const float*)d_in[19];
  const float* mod_b1 = (const float*)d_in[20];
  const float* mod_w2 = (const float*)d_in[21];
  const float* mod_b2 = (const float*)d_in[22];
  const float* pos_w  = (const float*)d_in[23];
  const float* pos_b  = (const float*)d_in[24];
  const float* dep_w1 = (const float*)d_in[25];
  const float* dep_b1 = (const float*)d_in[26];
  const float* dep_w2 = (const float*)d_in[27];
  const float* dep_b2 = (const float*)d_in[28];

  // Output regions (pred, decoded, world, occupancy, logits, heatmap)
  float* out = (float*)d_out;
  float* o_pred  = out;
  float* o_dec   = out + 12;
  float* o_world = o_dec + (size_t)4 * 8 * 128 * 4096;
  float* o_occ   = o_world + (size_t)4 * 8 * 128 * 4096;
  float* o_log   = o_occ + (size_t)4 * 8 * 4096;
  float* o_heat  = o_log + (size_t)4 * 4096;

  // Workspace carve-out (256B aligned)
  char* ws = (char*)d_ws;
  size_t o = 0;
  auto alloc = [&](size_t bytes) { size_t r = o; o = (o + bytes + 255) & ~(size_t)255; return r; };
  _Float16* xh   = (_Float16*)(ws + alloc((size_t)4 * 4096 * 256 * 2));     // 8 MiB
  float*    zr   = (float*)   (ws + alloc((size_t)4 * 256 * 4096 * 4));     // 16 MiB (htil reuse)
  float*    zbuf = (float*)   (ws + alloc((size_t)4 * 128 * 4096 * 4));     // 8 MiB
  float*    hbuf = (float*)   (ws + alloc((size_t)4 * 128 * 4096 * 4));     // 8 MiB
  _Float16* hs0  = (_Float16*)(ws + alloc((size_t)32 * 128 * 4096 * 2));    // 32 MiB
  _Float16* hs1  = (_Float16*)(ws + alloc((size_t)32 * 128 * 4096 * 2));    // 32 MiB
  _Float16* wzr0f = (_Float16*)(ws + alloc((size_t)16 * 9 * 8 * 512 * 2));
  _Float16* wh0f  = (_Float16*)(ws + alloc((size_t)8 * 9 * 8 * 512 * 2));
  _Float16* wzr1f = (_Float16*)(ws + alloc((size_t)16 * 9 * 8 * 512 * 2));
  _Float16* wh1f  = (_Float16*)(ws + alloc((size_t)8 * 9 * 8 * 512 * 2));
  _Float16* wpf   = (_Float16*)(ws + alloc((size_t)8 * 2 * 512 * 2));
  float*    cond  = (float*)(ws + alloc((size_t)4 * 128 * 4));
  float*    gb    = (float*)(ws + alloc((size_t)4 * 256 * 4));

  // 1) Swizzle weights to WMMA B-fragment layout (f16).
  k_prep_w3<<<2304, 256, 0, stream>>>(g0_wzr, wzr0f, 16);
  k_prep_w3<<<1152, 256, 0, stream>>>(g0_wh,  wh0f,  8);
  k_prep_w3<<<2304, 256, 0, stream>>>(g1_wzr, wzr1f, 16);
  k_prep_w3<<<1152, 256, 0, stream>>>(g1_wh,  wh1f,  8);
  k_prep_wp<<<32, 256, 0, stream>>>(patch_w, wpf);

  // 2) Patch embed + splat (world & occupancy live in d_out; zero them first).
  k_zero<<<1024, 256, 0, stream>>>(o_world, (size_t)4 * 8 * 128 * 4096 + (size_t)4 * 8 * 4096);
  k_patch<<<2048, 256, 0, stream>>>(obs, scoord, wpf, patch_b, o_world);
  k_occ<<<512, 256, 0, stream>>>(scoord, o_occ);
  k_norm_world<<<65536, 256, 0, stream>>>(o_world, o_occ);

  // 3) Sensor embeddings + FiLM modulation.
  k_sensor<<<4, 256, 0, stream>>>(scoord, sscale, s_w1, s_b1, s_w2, s_b2, cond, 0);
  k_sensor<<<4, 256, 0, stream>>>(tcoord, tscale, s_w1, s_b1, s_w2, s_b2, cond, 64);
  k_mod<<<4, 256, 0, stream>>>(cond, mod_w1, mod_b1, mod_w2, mod_b2, gb);

  // 4) Two ConvGRU layers over T=8 (the FLOP-dominant part, WMMA convs).
  for (int layer = 0; layer < 2; ++layer) {
    const _Float16* wzrf = layer ? wzr1f : wzr0f;
    const _Float16* whf  = layer ? wh1f  : wh0f;
    const float* bzr = layer ? g1_bzr : g0_bzr;
    const float* bh  = layer ? g1_bh  : g0_bh;
    _Float16* hs = layer ? hs1 : hs0;
    k_zero<<<1024, 256, 0, stream>>>(hbuf, (size_t)4 * 128 * 4096);
    for (int t = 0; t < 8; ++t) {
      if (layer == 0) k_pack_f32<<<16384, 256, 0, stream>>>(o_world, hbuf, xh, t);
      else            k_pack_f16<<<16384, 256, 0, stream>>>(hs0, hbuf, xh, t);
      k_conv3x3<16><<<256, 512, 0, stream>>>(xh, wzrf, bzr, zr);
      k_zr_act<<<8192, 256, 0, stream>>>(zr, hbuf, zbuf, xh);
      k_conv3x3<8><<<256, 512, 0, stream>>>(xh, whf, bh, zr /* htil reuse */);
      k_update<<<8192, 256, 0, stream>>>(zbuf, zr, hbuf, hs, t);
    }
  }

  // 5) Gather + FiLM -> decoded; heads.
  k_gather<<<65536, 256, 0, stream>>>(hs1, tcoord, gb, o_dec);
  k_logits<<<64, 256, 0, stream>>>(o_dec, pos_w, pos_b, o_log);
  k_head<<<4, 256, 0, stream>>>(o_log, o_dec, dep_w1, dep_b1, dep_w2, dep_b2, o_heat, o_pred);
}
